// LCA1_7352984011482
// MI455X (gfx1250) — matrix operand order repaired and checked by reading, not compiled, not run on hardware
//
#include <hip/hip_runtime.h>
#include <hip/hip_bf16.h>

// Fused LCA kernel for MI455X (gfx1250, wave32).
// One workgroup = one batch image x one 16x16 pixel tile x all 32 channels.
// Pipeline (all in LDS, single HBM read of cen + single HBM write of out):
//   P0: stage cen 24x24 halo tile (zero-padded at image borders) + weights
//   P1: spatial attention on 18x18 region (dilations 1 and 3)
//   P2: depthwise 3x3 conv -> g[32][256]
//   P3: pointwise 1x1 conv as GEMM via v_wmma_f32_16x16x32_bf16 (K=32 exact),
//       fast sigmoid (v_exp + v_rcp), out = cen*sig + cen, coalesced stores.

typedef __attribute__((ext_vector_type(16))) __bf16 v16bf;
typedef __attribute__((ext_vector_type(8)))  float  v8f;

#define TILE   16
#define CH     32
#define HW     256
#define CEN_CH 580   // padded per-channel stride for 24x24 tile (580%64=4 -> conflict-free channel-split reads)
#define OUT_CH 324   // 18*18 attention region
#define G_STR  257   // 8*257 % 64 == 8 -> lane halves (8 ch apart) share 8 banks
#define PW_STR 33    // odd stride; A-fragment lane halves (8 K apart) share 8 banks

__device__ __forceinline__ float fast_sigmoid(float x) {
    // v_exp_f32 + v_rcp_f32; avoids IEEE div-scale fixup chain
    return __builtin_amdgcn_rcpf(1.0f + __expf(-x));
}

__global__ __launch_bounds__(256)
void lca_fused_gfx1250(const float* __restrict__ cen,
                       const float* __restrict__ dw,
                       const float* __restrict__ pw,
                       float* __restrict__ out)
{
    __shared__ float s_cen[CH * CEN_CH];  //  74,240 B
    __shared__ float s_att[CH * OUT_CH];  //  41,472 B
    __shared__ float s_g  [CH * G_STR];   //  32,896 B
    __shared__ float s_dw [CH * 9];       //   1,152 B
    __shared__ float s_pw [CH * PW_STR];  //   4,224 B

    const int tid = threadIdx.x;
    const int x0  = blockIdx.x * TILE;
    const int y0  = blockIdx.y * TILE;
    const int b   = blockIdx.z;

    // ---------------- Phase 0: weights + cen halo tile ----------------
    if (tid < CH * 9) s_dw[tid] = dw[tid];
    for (int i = tid; i < CH * CH; i += 256)
        s_pw[(i >> 5) * PW_STR + (i & 31)] = pw[i];

    const float* cen_b = cen + (size_t)b * CH * HW * HW;
    {
        // Each thread owns up to 3 fixed spatial positions of the 24x24 tile,
        // invariant across channels: precompute coords once, then sweep
        // channels with incremental pointer adds (no per-iteration div/mod).
#pragma unroll
        for (int s = 0; s < 3; ++s) {
            const int j = tid + s * 256;
            if (j < 576) {
                const int r  = j / 24;
                const int c  = j - r * 24;
                const int gy = y0 - 4 + r;
                const int gx = x0 - 4 + c;
                const bool inb = (gy >= 0) & (gy < HW) & (gx >= 0) & (gx < HW);
                const float* src = cen_b + ((size_t)(inb ? gy : 0) * HW + (inb ? gx : 0));
                float* dst = s_cen + r * 24 + c;
                for (int ch = 0; ch < CH; ++ch) {
                    float v = inb ? src[(size_t)ch * HW * HW] : 0.0f; // SAME-pad zeros
                    dst[ch * CEN_CH] = v;
                }
            }
        }
    }
    __syncthreads();

    // ---------------- Phase 1: spatial attention (18x18 region) -------
    {
#pragma unroll
        for (int s2 = 0; s2 < 2; ++s2) {
            const int j = tid + s2 * 256;
            if (j < OUT_CH) {
                const int r  = j / 18;
                const int c  = j - r * 18;
                const int gy = y0 - 1 + r, gx = x0 - 1 + c;
                const bool inside = (gy >= 0) & (gy < HW) & (gx >= 0) & (gx < HW);
                const float* base0 = s_cen + (r + 3) * 24 + (c + 3);
                float* dst = s_att + j;
                for (int ch = 0; ch < CH; ++ch) {
                    const float* base = base0 + ch * CEN_CH;
                    float ctr = base[0];
                    float m[2];
#pragma unroll
                    for (int s = 0; s < 2; ++s) {
                        const int d = (s == 0) ? 1 : 3;
                        const int v = d * 24;
                        float p0 = (ctr - base[-v - d]) * (ctr - base[ v + d]);  // diag
                        float p1 = (ctr - base[-v    ]) * (ctr - base[ v    ]);  // vert
                        float p2 = (ctr - base[-v + d]) * (ctr - base[ v - d]);  // anti-diag
                        float p3 = (ctr - base[    -d]) * (ctr - base[     d]);  // horiz
                        m[s] = fminf(fminf(p0, p1), fminf(p2, p3));
                    }
                    float o = fmaxf(m[0], m[1]) + 0.5f * (m[0] + m[1]);
                    dst[ch * OUT_CH] = inside ? o : 0.0f;  // SAME pad of attention map
                }
            }
        }
    }
    __syncthreads();

    // ---------------- Phase 2: depthwise 3x3 -> g ---------------------
    {
        const int px = tid & 15, py = tid >> 4;   // 256 threads == 256 pixels
        const float* ob = s_att + py * 18 + px;
        const float* w  = s_dw;
        float* gdst = s_g + tid;
        for (int ch = 0; ch < CH; ++ch) {
            float acc = 0.0f;
#pragma unroll
            for (int ky = 0; ky < 3; ++ky)
#pragma unroll
                for (int kx = 0; kx < 3; ++kx)
                    acc = fmaf(w[ky * 3 + kx], ob[ky * 18 + kx], acc);
            gdst[0] = acc;
            ob += OUT_CH; w += 9; gdst += G_STR;
        }
    }
    __syncthreads();

    // ---------------- Phase 3: 1x1 conv via WMMA + epilogue -----------
    const int wave = tid >> 5;
    const int lane = tid & 31;
    const int half = lane >> 4;     // lane half selects K split (ISA 7.12.2)
    const int mrow = lane & 15;

    // A fragments: pw_weight rows (out-channel = M), K = in-channel.
    // 16-bit A layout: lanes 0-15 -> K{0..7,16..23}; lanes 16-31 -> K{8..15,24..31}
    v16bf a0, a1;
#pragma unroll
    for (int e = 0; e < 16; ++e) {
        int ci = e + ((e >= 8) ? 8 : 0) + half * 8;
        a0[e] = (__bf16)s_pw[ mrow       * PW_STR + ci];
        a1[e] = (__bf16)s_pw[(mrow + 16) * PW_STR + ci];
    }

    float* out_b = out + (size_t)b * CH * HW * HW;
#pragma unroll
    for (int t = 0; t < 2; ++t) {
        const int pg = wave * 2 + t;         // pixel group == tile row (width 16)
        const int p  = pg * 16 + mrow;

        // B fragment: g channels (K) x pixels (N); lanes 0-15 K{0..7,16..23}, lanes 16-31 K{8..15,24..31}
        v16bf bf;
#pragma unroll
        for (int e = 0; e < 16; ++e)
            bf[e] = (__bf16)s_g[(e + ((e >= 8) ? 8 : 0) + half * 8) * G_STR + p];

        v8f cz = {};
        v8f d0 = __builtin_amdgcn_wmma_f32_16x16x32_bf16(
            false, a0, false, bf, (short)0, cz, false, false);
        v8f d1 = __builtin_amdgcn_wmma_f32_16x16x32_bf16(
            false, a1, false, bf, (short)0, cz, false, false);

        // D layout: VGPR i -> M=i (lanes 0-15) / M=i+8 (lanes 16-31).
        // Common per-lane base pointers; channel strides become immediate
        // offsets (DS: i*CEN_CH*4 <= 34.8KB < 64KB; VMEM: i*256KB <= 1.8MB < 8MB).
        const int py = pg, px = mrow;
        const int gy = y0 + py, gx = x0 + px;
        const float* cb   = s_cen + (half * 8) * CEN_CH + (py + 4) * 24 + (px + 4);
        const float* cbHi = cb + 16 * CEN_CH;
        float* pb   = out_b + (size_t)(half * 8) * (HW * HW) + (size_t)gy * HW + gx;
        float* pbHi = pb + (size_t)16 * (HW * HW);
#pragma unroll
        for (int i = 0; i < 8; ++i) {
            float c0 = cb  [i * CEN_CH];
            float c1 = cbHi[i * CEN_CH];
            float s0 = fast_sigmoid(d0[i]);
            float s1 = fast_sigmoid(d1[i]);
            pb  [i * (HW * HW)] = fmaf(c0, s0, c0);
            pbHi[i * (HW * HW)] = fmaf(c1, s1, c1);
        }
    }
}

extern "C" void kernel_launch(void* const* d_in, const int* in_sizes, int n_in,
                              void* d_out, int out_size, void* d_ws, size_t ws_size,
                              hipStream_t stream) {
    const float* cen = (const float*)d_in[0];
    // d_in[1] (mask) is unused by the reference computation.
    const float* dw  = (const float*)d_in[2];
    const float* pw  = (const float*)d_in[3];
    float* out = (float*)d_out;

    const int B = in_sizes[0] / (CH * HW * HW);   // 8 for the reference shapes
    dim3 grid(HW / TILE, HW / TILE, B);
    lca_fused_gfx1250<<<grid, 256, 0, stream>>>(cen, dw, pw, out);
}
